// AttentionLayer_28003186770560
// MI455X (gfx1250) — matrix operand order repaired and checked by reading, not compiled, not run on hardware
//
#include <hip/hip_runtime.h>

typedef __bf16 bf16_t;
typedef __attribute__((ext_vector_type(16))) __bf16 v16bf;
typedef __attribute__((ext_vector_type(8)))  __bf16 v8bf;
typedef __attribute__((ext_vector_type(8)))  float   v8f;
typedef __attribute__((ext_vector_type(4)))  unsigned int u32x4;
typedef __attribute__((ext_vector_type(8)))  int i32x8;
typedef __attribute__((ext_vector_type(4)))  int i32x4;

namespace {
constexpr int Bn = 8, Ln = 1024, En = 1024, Hn = 16, Dn = 64;
constexpr int Fn = 3 * En;      // 3072 packed QKV columns
constexpr int Mn = Bn * Ln;     // 8192 rows
constexpr float kEps    = 1e-5f;
constexpr float kQScale = 0.125f;                     // 1/sqrt(D)
constexpr float kAttnSc = 1.0f / (Hn * (float)Ln);    // head+query average
}

#if __has_builtin(__builtin_amdgcn_tensor_load_to_lds)
#define HAVE_TDM 1
#else
#define HAVE_TDM 0
#endif

// ---- WMMA helpers (CDNA5 wave32 fragment layouts, CDNA5 ISA) ----

// A-matrix 16x32 bf16: lane L holds row M=L%16.
// lanes 0-15:  K = k0..k0+15
// lanes 16-31: K = k0+16..k0+31 (contiguous, matches B)
__device__ __forceinline__ v16bf load_a16(const bf16_t* rowp_k0, int lane) {
  union { v16bf v; v8bf h[2]; } u;
  const int lo = (lane & 16) ? 16 : 0;
  u.h[0] = *(const v8bf*)(rowp_k0 + lo);
  u.h[1] = *(const v8bf*)(rowp_k0 + lo + 8);
  return u.v;
}

// B-matrix 32x16 bf16: lane L holds column N=L%16;
// lanes 0-15 hold K=k0..k0+15, lanes 16-31 hold K=k0+16..k0+31 (contiguous).
__device__ __forceinline__ v16bf load_b16(const bf16_t* colp) {
  return *(const v16bf*)colp;
}

__device__ __forceinline__ v8f wmma_bf16(v16bf a, v16bf b, v8f c) {
  return __builtin_amdgcn_wmma_f32_16x16x32_bf16(
      /*neg_a=*/false, a, /*neg_b=*/false, b,
      /*c_mod=*/(short)0, c, /*reuse_a=*/false, /*reuse_b=*/false);
}

// ---- Tensor Data Mover: 2-D tile load, 2-byte elements (CDNA5 ISA) -----
// Loads a 64-row x 64-col bf16 tile (row stride = stride_elems) into LDS,
// rows packed contiguously (64*2 = 128 B apart). Tracked by TENSORcnt.
#if HAVE_TDM
__device__ __forceinline__ void tdm_load_tile64x64(unsigned int lds_off,
                                                   const bf16_t* gsrc,
                                                   unsigned int stride_elems) {
  const unsigned long long ga = (unsigned long long)(uintptr_t)gsrc;
  u32x4 g0;
  g0[0] = 1u;                                          // count=1 (valid D#)
  g0[1] = lds_off;                                     // lds_addr (bytes)
  g0[2] = (unsigned int)ga;                            // global_addr[31:0]
  g0[3] = (unsigned int)((ga >> 32) & 0x1FFFFFFu)      // global_addr[56:32]
          | (2u << 30);                                // type = 2 ("image")
  i32x8 g1;
  g1[0] = (1 << 16);                                   // mask=0, data_size=2B
  g1[1] = (int)((0x100000u & 0xFFFFu) << 16);          // tensor_dim0 low16
  g1[2] = (int)(((0x100000u >> 16) & 0xFFFFu)          // tensor_dim0 hi16
          | ((0x100000u & 0xFFFFu) << 16));            // tensor_dim1 low16
  g1[3] = (int)(((0x100000u >> 16) & 0xFFFFu)          // tensor_dim1 hi16
          | (64u << 16));                              // tile_dim0 = 64
  g1[4] = 64;                                          // tile_dim1=64, tile_dim2=0
  g1[5] = (int)stride_elems;                           // tensor_dim0_stride lo32
  g1[6] = 0;                                           // stride hi16 | dim1_stride
  g1[7] = 0;
  const i32x4 z4 = {0, 0, 0, 0};
#if defined(__clang_major__) && __clang_major__ >= 23
  const i32x8 z8 = {0, 0, 0, 0, 0, 0, 0, 0};
  __builtin_amdgcn_tensor_load_to_lds(g0, g1, z4, z4, z8, 0);
#else
  __builtin_amdgcn_tensor_load_to_lds(g0, g1, z4, z4, 0);
#endif
}
#endif

// ------------------------------- utility kernels ---------------------------

__global__ __launch_bounds__(256) void zero_f32(float* p, int n) {
  int i = blockIdx.x * blockDim.x + threadIdx.x;
  int st = gridDim.x * blockDim.x;
  for (; i < n; i += st) p[i] = 0.0f;
}

__global__ __launch_bounds__(256) void f32_to_bf16(const float* __restrict__ s,
                                                   bf16_t* __restrict__ d, int n) {
  int i = blockIdx.x * blockDim.x + threadIdx.x;
  int st = gridDim.x * blockDim.x;
  for (; i < n; i += st) d[i] = (bf16_t)s[i];
}

// ------------------- QKV projection: TDM-staged tiled GEMM -----------------
// out[m,f] = sum_e x[m,e] * w_qkv[f,e] + b_qkv[f]; scatter into q/k/vT.
// Workgroup = 8 waves computes a 64x64 output tile; A/B K-slabs (64x64 bf16,
// 8 KB each) staged in LDS by the Tensor Data Mover, double buffered so the
// DMA of chunk k+1 overlaps WMMA on chunk k.
__global__ __launch_bounds__(256) void qkv_gemm(const bf16_t* __restrict__ xbf,
                                                const bf16_t* __restrict__ wbf,
                                                const float* __restrict__ bias,
                                                bf16_t* __restrict__ qbf,
                                                bf16_t* __restrict__ kbf,
                                                bf16_t* __restrict__ vT) {
  __shared__ bf16_t As[2][64 * 64];   // 2 x 8 KB
  __shared__ bf16_t Bs[2][64 * 64];   // 2 x 8 KB
  const int m0 = blockIdx.x * 64;     // output rows   (x rows)
  const int n0 = blockIdx.y * 64;     // output cols   (w_qkv rows)
  const int tid = threadIdx.x, wave = tid >> 5, lane = tid & 31;
  const int mi  = wave >> 1;                   // 0..3: 16-row stripe
  const int ni0 = (wave & 1) * 2;              // two 16-col stripes
  constexpr int NC = En / 64;                  // 16 K-chunks

  v8f c0 = {}, c1 = {};

#if HAVE_TDM
  if (wave == 0) {   // chunk 0 prefetch (whole-wave uniform branch)
    tdm_load_tile64x64((unsigned int)(uintptr_t)&As[0][0],
                       xbf + (size_t)m0 * En, En);
    tdm_load_tile64x64((unsigned int)(uintptr_t)&Bs[0][0],
                       wbf + (size_t)n0 * En, En);
  }
#endif

  for (int chunk = 0; chunk < NC; ++chunk) {
#if HAVE_TDM
    const int cur = chunk & 1;
    if (wave == 0) {
      if (chunk + 1 < NC) {          // prefetch next chunk into other buffer
        const int nxt = cur ^ 1;
        tdm_load_tile64x64((unsigned int)(uintptr_t)&As[nxt][0],
                           xbf + (size_t)m0 * En + (chunk + 1) * 64, En);
        tdm_load_tile64x64((unsigned int)(uintptr_t)&Bs[nxt][0],
                           wbf + (size_t)n0 * En + (chunk + 1) * 64, En);
#if __has_builtin(__builtin_amdgcn_s_wait_tensorcnt)
        __builtin_amdgcn_s_wait_tensorcnt(2);  // current chunk complete
      } else {
        __builtin_amdgcn_s_wait_tensorcnt(0);  // drain
#endif
      }
    }
    __syncthreads();                 // publish staged data to all waves
#else
    const int cur = 0;
    for (int i = tid; i < 64 * 64; i += 256) {
      const int r = i >> 6, cc = i & 63;
      As[0][i] = xbf[(size_t)(m0 + r) * En + chunk * 64 + cc];
      Bs[0][i] = wbf[(size_t)(n0 + r) * En + chunk * 64 + cc];
    }
    __syncthreads();
#endif

    const bf16_t* arow = &As[cur][(mi * 16 + (lane & 15)) * 64];
    const bf16_t* bc0  = &Bs[cur][((ni0 + 0) * 16 + (lane & 15)) * 64 + ((lane >> 4) * 16)];
    const bf16_t* bc1  = &Bs[cur][((ni0 + 1) * 16 + (lane & 15)) * 64 + ((lane >> 4) * 16)];
    for (int k0 = 0; k0 < 64; k0 += 32) {
      const v16bf a = load_a16(arow + k0, lane);
      c0 = wmma_bf16(a, load_b16(bc0 + k0), c0);
      c1 = wmma_bf16(a, load_b16(bc1 + k0), c1);
    }
    __syncthreads();                 // all waves done before buffer is reused
  }

  // Epilogue: bias + scatter into q (scaled), k, and transposed v
  for (int t = 0; t < 2; ++t) {
    const v8f& c = t ? c1 : c0;
    const int n = n0 + (ni0 + t) * 16 + (lane & 15);
    const float bia = bias[n];
    for (int r = 0; r < 8; ++r) {
      const int m    = m0 + mi * 16 + r + ((lane >> 4) << 3);
      const int bidx = m / Ln, l = m % Ln;
      const float val = c[r] + bia;
      if (n < En) {                                // Q (pre-scaled)
        const int hh = n / Dn, dd = n % Dn;
        qbf[(((size_t)bidx * Hn + hh) * Ln + l) * Dn + dd] = (bf16_t)(val * kQScale);
      } else if (n < 2 * En) {                     // K
        const int f = n - En, hh = f / Dn, dd = f % Dn;
        kbf[(((size_t)bidx * Hn + hh) * Ln + l) * Dn + dd] = (bf16_t)val;
      } else {                                     // V, stored [b][h][d][l]
        const int f = n - 2 * En, hh = f / Dn, dd = f % Dn;
        vT[(((size_t)bidx * Hn + hh) * Dn + dd) * Ln + l] = (bf16_t)val;
      }
    }
  }
}

// --------------- attention: scores -> softmax -> avg_attn -> ctx -----------
__global__ __launch_bounds__(128) void attn_kernel(const bf16_t* __restrict__ qbf,
                                                   const bf16_t* __restrict__ kbf,
                                                   const bf16_t* __restrict__ vT,
                                                   bf16_t* __restrict__ ctxbf,
                                                   float* __restrict__ avg_attn) {
  __shared__ float S[16][Ln];                      // 64 KB of the 320 KB WGP LDS
  const int bidx = blockIdx.x / Hn, hh = blockIdx.x % Hn;
  const int l0 = blockIdx.y * 16;
  const int tid = threadIdx.x, wave = tid >> 5, lane = tid & 31;

  const bf16_t* qbase = qbf + (((size_t)bidx * Hn + hh) * Ln + l0) * Dn;
  const bf16_t* kbase = kbf + ((size_t)bidx * Hn + hh) * Ln * Dn;
  const bf16_t* vbase = vT  + ((size_t)bidx * Hn + hh) * Dn * Ln;

  // Phase 1: scores tile 16 x 1024 (K-dim = D = 64 -> 2 WMMAs per 16x16 tile)
  for (int nt = wave; nt < Ln / 16; nt += 4) {
    const int n0 = nt * 16;
    v8f c = {};
    const bf16_t* ar = qbase + (size_t)(lane & 15) * Dn;
    const bf16_t* br = kbase + (size_t)(n0 + (lane & 15)) * Dn + ((lane >> 4) * 16);
    for (int k0 = 0; k0 < Dn; k0 += 32)
      c = wmma_bf16(load_a16(ar + k0, lane), load_b16(br + k0), c);
    const int n = n0 + (lane & 15);
    for (int r = 0; r < 8; ++r) S[r + ((lane >> 4) << 3)][n] = c[r];
  }
  __syncthreads();

  // Phase 2: softmax, one wave per 4 rows, wave32 shuffle reductions
  for (int rr = 0; rr < 4; ++rr) {
    const int m = wave * 4 + rr;
    float mx = -1e30f;
    for (int cc = lane; cc < Ln; cc += 32) mx = fmaxf(mx, S[m][cc]);
    for (int o = 16; o; o >>= 1) mx = fmaxf(mx, __shfl_xor(mx, o, 32));
    float sum = 0.0f;
    for (int cc = lane; cc < Ln; cc += 32) {
      const float e = expf(S[m][cc] - mx);
      S[m][cc] = e;
      sum += e;
    }
    for (int o = 16; o; o >>= 1) sum += __shfl_xor(sum, o, 32);
    const float inv = 1.0f / sum;
    for (int cc = lane; cc < Ln; cc += 32) S[m][cc] *= inv;
  }
  __syncthreads();

  // avg_attn[b,m] += (1/(H*L)) * sum over 16 queries of P[l, m]
  for (int cc = tid; cc < Ln; cc += 128) {
    float s = 0.0f;
    for (int m = 0; m < 16; ++m) s += S[m][cc];
    atomicAdd(&avg_attn[(size_t)bidx * Ln + cc], s * kAttnSc);
  }

  // Phase 3: ctx = P(16x1024) x V(1024x64); each wave owns a 16-wide d chunk
  const int d0 = wave * 16;
  v8f c = {};
  for (int k0 = 0; k0 < Ln; k0 += 32) {
    union { v16bf v; bf16_t e[16]; } a;
    const int m  = lane & 15;
    const int b0 = k0 + ((lane & 16) ? 16 : 0);
    for (int j = 0; j < 8; ++j) {
      a.e[j]     = (bf16_t)S[m][b0 + j];
      a.e[8 + j] = (bf16_t)S[m][b0 + 8 + j];
    }
    const bf16_t* br = vbase + (size_t)(d0 + (lane & 15)) * Ln + k0 + ((lane >> 4) * 16);
    c = wmma_bf16(a.v, load_b16(br), c);
  }
  const int n = d0 + (lane & 15);
  for (int r = 0; r < 8; ++r) {
    const int m = r + ((lane >> 4) << 3);
    ctxbf[(size_t)(bidx * Ln + l0 + m) * En + hh * Dn + n] = (bf16_t)c[r];
  }
}

// --------- output projection + residual + LayerNorm + mean-pool ------------
__global__ __launch_bounds__(256) void outproj_ln(const bf16_t* __restrict__ ctxbf,
                                                  const bf16_t* __restrict__ wobf,
                                                  const float* __restrict__ bout,
                                                  const float* __restrict__ x,
                                                  const float* __restrict__ gamma,
                                                  const float* __restrict__ beta,
                                                  float* __restrict__ pooled) {
  __shared__ float O[16][En];                      // 64 KB stripe
  const int m0 = blockIdx.x * 16;
  const int bidx = m0 / Ln;
  const int tid = threadIdx.x, wave = tid >> 5, lane = tid & 31;

  // GEMM: each of 8 waves computes 8 16x16 tiles (128 output columns)
  for (int nt = 0; nt < 8; ++nt) {
    const int n0 = wave * 128 + nt * 16;
    const bf16_t* arow = ctxbf + (size_t)(m0 + (lane & 15)) * En;
    const bf16_t* bcol = wobf + (size_t)(n0 + (lane & 15)) * En + ((lane >> 4) * 16);
    v8f c = {};
    for (int k0 = 0; k0 < En; k0 += 32)
      c = wmma_bf16(load_a16(arow + k0, lane), load_b16(bcol + k0), c);
    const int n = n0 + (lane & 15);
    const float bia = bout[n];
    for (int r = 0; r < 8; ++r) O[r + ((lane >> 4) << 3)][n] = c[r] + bia;
  }
  __syncthreads();

  // residual + LayerNorm + pooled accumulation: 8 waves x 2 rows
  for (int rr = 0; rr < 2; ++rr) {
    const int m = wave * 2 + rr;
    const float* xrow = x + (size_t)(m0 + m) * En;
    float sum = 0.0f;
    for (int cc = lane; cc < En; cc += 32) {
      const float hv = xrow[cc] + O[m][cc];
      O[m][cc] = hv;
      sum += hv;
    }
    for (int o = 16; o; o >>= 1) sum += __shfl_xor(sum, o, 32);
    const float mu = sum * (1.0f / En);
    float sq = 0.0f;
    for (int cc = lane; cc < En; cc += 32) {
      const float dv = O[m][cc] - mu;
      sq += dv * dv;
    }
    for (int o = 16; o; o >>= 1) sq += __shfl_xor(sq, o, 32);
    const float rstd = rsqrtf(sq * (1.0f / En) + kEps);
    for (int cc = lane; cc < En; cc += 32) {
      const float hn = (O[m][cc] - mu) * rstd * gamma[cc] + beta[cc];
      atomicAdd(&pooled[(size_t)bidx * En + cc], hn * (1.0f / Ln));
    }
  }
}

// ------------------------------- launcher ----------------------------------
extern "C" void kernel_launch(void* const* d_in, const int* in_sizes, int n_in,
                              void* d_out, int out_size, void* d_ws, size_t ws_size,
                              hipStream_t stream) {
  (void)in_sizes; (void)n_in; (void)out_size; (void)ws_size;
  const float* x     = (const float*)d_in[0];
  const float* w_qkv = (const float*)d_in[1];
  const float* b_qkv = (const float*)d_in[2];
  const float* w_out = (const float*)d_in[3];
  const float* b_out = (const float*)d_in[4];
  const float* gamma = (const float*)d_in[5];
  const float* beta  = (const float*)d_in[6];
  float* out = (float*)d_out;

  char* ws = (char*)d_ws;
  size_t off = 0;
  auto carve = [&](size_t bytes) -> char* {
    char* p = ws + off;
    off += (bytes + 255) & ~(size_t)255;
    return p;
  };
  bf16_t* xbf   = (bf16_t*)carve((size_t)Mn * En * 2);        // 16 MB
  bf16_t* wqbf  = (bf16_t*)carve((size_t)Fn * En * 2);        //  6 MB
  bf16_t* wobf  = (bf16_t*)carve((size_t)En * En * 2);        //  2 MB
  bf16_t* qbf   = (bf16_t*)carve((size_t)Bn * Hn * Ln * Dn * 2);
  bf16_t* kbf   = (bf16_t*)carve((size_t)Bn * Hn * Ln * Dn * 2);
  bf16_t* vT    = (bf16_t*)carve((size_t)Bn * Hn * Dn * Ln * 2);
  bf16_t* ctxbf = (bf16_t*)carve((size_t)Mn * En * 2);        // total ~88 MB

  zero_f32<<<dim3(64), dim3(256), 0, stream>>>(out, Bn * En + Bn * Ln);
  f32_to_bf16<<<dim3(4096), dim3(256), 0, stream>>>(x, xbf, Mn * En);
  f32_to_bf16<<<dim3(4096), dim3(256), 0, stream>>>(w_qkv, wqbf, Fn * En);
  f32_to_bf16<<<dim3(2048), dim3(256), 0, stream>>>(w_out, wobf, En * En);

  qkv_gemm<<<dim3(Mn / 64, Fn / 64), dim3(256), 0, stream>>>(xbf, wqbf, b_qkv,
                                                             qbf, kbf, vT);
  attn_kernel<<<dim3(Bn * Hn, Ln / 16), dim3(128), 0, stream>>>(qbf, kbf, vT,
                                                                ctxbf,
                                                                out + Bn * En);
  outproj_ln<<<dim3(Mn / 16), dim3(256), 0, stream>>>(ctxbf, wobf, b_out, x,
                                                      gamma, beta, out);
}
